// AttentionBlock_19112604467810
// MI455X (gfx1250) — matrix operand order repaired and checked by reading, not compiled, not run on hardware
//
#include <hip/hip_runtime.h>

typedef __attribute__((ext_vector_type(16))) __bf16 v16bf;
typedef __attribute__((ext_vector_type(8)))  float  v8f;
typedef unsigned int u32x4 __attribute__((ext_vector_type(4)));
typedef int          i32x8 __attribute__((ext_vector_type(8)));
typedef int          i32x4 __attribute__((ext_vector_type(4)));

union AFrag { uint4 u[2]; v16bf v; };

#define WMMA_BF16(a, b, c) \
    __builtin_amdgcn_wmma_f32_16x16x32_bf16(false, (a), false, (b), (short)0, (c), false, false)

#if defined(__has_builtin)
#if __has_builtin(__builtin_amdgcn_tensor_load_to_lds) && __has_builtin(__builtin_amdgcn_s_wait_tensorcnt)
#define USE_TDM 1
#endif
#endif
#ifndef USE_TDM
#define USE_TDM 0
#endif

__device__ __forceinline__ unsigned short f2bf(float f) {
    unsigned int u = __float_as_uint(f);
    return (unsigned short)((u + 0x7FFFu + ((u >> 16) & 1u)) >> 16);
}

// packed f32x2 -> bf16x2 in one native VOP3 op
__device__ __forceinline__ unsigned pk_bf16(float lo, float hi) {
    unsigned r;
    asm("v_cvt_pk_bf16_f32 %0, %1, %2" : "=v"(r) : "v"(lo), "v"(hi));
    return r;
}

// ---------------------------------------------------------------------------
// Kernel 1: QKV projection. X[8192,1024] @ W[1024,512] + b -> bf16.
// z=0 -> Q row-major [8192,512], z=1 -> K row-major, z=2 -> V transposed [4,512,2048]
// Block tile 64(M) x 128(N), 8 waves (4x2), each wave 16x64 via 4 WMMA accums.
// ---------------------------------------------------------------------------
__launch_bounds__(256)
__global__ void qkv_proj_kernel(const float* __restrict__ X,
                                const float* __restrict__ Wq, const float* __restrict__ bq,
                                const float* __restrict__ Wk, const float* __restrict__ bk,
                                const float* __restrict__ Wv, const float* __restrict__ bv,
                                unsigned short* __restrict__ qo,
                                unsigned short* __restrict__ ko,
                                unsigned short* __restrict__ vto) {
    const int z = blockIdx.z;
    const float* W    = (z == 0) ? Wq : ((z == 1) ? Wk : Wv);
    const float* bias = (z == 0) ? bq : ((z == 1) ? bk : bv);

    const int n0 = blockIdx.x * 128;
    const int m0 = blockIdx.y * 64;

    __shared__ __align__(16) unsigned short a_lds[64][40];    // X tile bf16, padded
    __shared__ __align__(16) unsigned short wt_lds[128][40];  // W^T tile [n][k], padded

    const int tid  = threadIdx.x;
    const int lane = tid & 31;
    const int wave = tid >> 5;
    const int wm   = wave & 3;
    const int wn   = wave >> 2;
    const int half = lane >> 4;
    const int l16  = lane & 15;

    v8f acc[4];
#pragma unroll
    for (int j = 0; j < 4; j++)
#pragma unroll
        for (int e = 0; e < 8; e++) acc[j][e] = 0.0f;

    for (int k0 = 0; k0 < 1024; k0 += 32) {
        // stage A: 64x32, packed-pair convert
#pragma unroll
        for (int i = 0; i < 4; i++) {
            int e = tid + i * 256;
            int r = e >> 4, c2 = e & 15;
            const float* xp = &X[(m0 + r) * 1024 + k0 + c2 * 2];
            *(unsigned*)&a_lds[r][c2 * 2] = pk_bf16(xp[0], xp[1]);
        }
        // stage W transposed: wt[n][k] pairs along k
#pragma unroll
        for (int i = 0; i < 8; i++) {
            int e = tid + i * 256;
            int n = e & 127, kp = e >> 7;
            float w0 = W[(k0 + 2 * kp) * 512 + n0 + n];
            float w1 = W[(k0 + 2 * kp + 1) * 512 + n0 + n];
            *(unsigned*)&wt_lds[n][kp * 2] = pk_bf16(w0, w1);
        }
        __syncthreads();

        AFrag a;
        a.u[0] = *(const uint4*)&a_lds[wm * 16 + l16][half * 8];
        a.u[1] = *(const uint4*)&a_lds[wm * 16 + l16][16 + half * 8];
#pragma unroll
        for (int j = 0; j < 4; j++) {
            AFrag bf;
            const unsigned short* wrow = wt_lds[wn * 64 + j * 16 + l16];
            bf.u[0] = *(const uint4*)&wrow[half * 8];
            bf.u[1] = *(const uint4*)&wrow[16 + half * 8];
            acc[j] = WMMA_BF16(a.v, bf.v, acc[j]);
        }
        __syncthreads();
    }

#pragma unroll
    for (int j = 0; j < 4; j++) {
        int n = n0 + wn * 64 + j * 16 + l16;
        float bb = bias[n];
#pragma unroll
        for (int r = 0; r < 8; r += 2) {
            unsigned pkv = pk_bf16(acc[j][r] + bb, acc[j][r + 1] + bb);
            int m = m0 + wm * 16 + r + half * 8;   // rows m and m+1
            if (z == 0) {
                qo[m * 512 + n]       = (unsigned short)pkv;
                qo[(m + 1) * 512 + n] = (unsigned short)(pkv >> 16);
            } else if (z == 1) {
                ko[m * 512 + n]       = (unsigned short)pkv;
                ko[(m + 1) * 512 + n] = (unsigned short)(pkv >> 16);
            } else {
                int bidx = m >> 11, t = m & 2047;
                vto[((bidx << 9) + n) * 2048 + t]     = (unsigned short)pkv;
                vto[((bidx << 9) + n) * 2048 + t + 1] = (unsigned short)(pkv >> 16);
            }
        }
    }
}

// ---------------------------------------------------------------------------
// Kernel 2: copy minibatch into out[:, 0:1024] (row stride 1536)
// ---------------------------------------------------------------------------
__launch_bounds__(256)
__global__ void copy_passthrough(const float* __restrict__ in, float* __restrict__ out) {
    int i = blockIdx.x * 256 + threadIdx.x;
    int row = i >> 8;
    int col = (i & 255) << 2;
    float4 v = *(const float4*)&in[row * 1024 + col];
    *(float4*)&out[row * 1536 + col] = v;
}

// ---------------------------------------------------------------------------
// Kernel 3: fused causal attention, flash-style online softmax.
// Workgroup = 8 waves covering 128 query rows; uniform key-chunk loop.
// K chunks (64 keys x 512 bf16) staged into double-buffered LDS by the
// Tensor Data Mover (hardware row padding), overlapped with compute.
// ---------------------------------------------------------------------------
#define KPAD  520   // 512 + 8 shorts pad per key row (1040B, 16B aligned)
#define CHUNK 64    // keys per staged chunk

__launch_bounds__(256)
__global__ void attn_kernel(const unsigned short* __restrict__ q,   // [8192,512] bf16
                            const unsigned short* __restrict__ k,   // [8192,512] bf16
                            const unsigned short* __restrict__ vt,  // [4,512,2048] bf16
                            float* __restrict__ out) {              // [8192,1536] f32
    const int tid    = threadIdx.x;
    const int lane   = tid & 31;
    const int wave   = tid >> 5;
    const int half   = lane >> 4;
    const int l16    = lane & 15;
    const int b      = blockIdx.y;
    const int t0_blk = blockIdx.x * 128;
    const int t0     = t0_blk + wave * 16;
    const int row_base = b * 2048 + t0;

    __shared__ __align__(16) unsigned short kc_lds[2][CHUNK * KPAD]; // double-buffered K
    __shared__ __align__(16) unsigned short p_lds[8][16][CHUNK];     // per-wave P tile

    const unsigned short* kbase = k + (long)b * 2048 * 512;
    const unsigned short* vbase = vt + (long)b * 512 * 2048;

    // Preload Q fragments for the 16x512 tile (16 fragments, k-dim 32 each)
    AFrag qf[16];
    {
        const unsigned short* qrow = q + (long)(row_base + l16) * 512;
#pragma unroll
        for (int kk = 0; kk < 16; kk++) {
            qf[kk].u[0] = *(const uint4*)&qrow[kk * 32 + half * 8];
            qf[kk].u[1] = *(const uint4*)&qrow[kk * 32 + 16 + half * 8];
        }
    }

    v8f o[32];
#pragma unroll
    for (int j = 0; j < 32; j++)
#pragma unroll
        for (int e = 0; e < 8; e++) o[j][e] = 0.0f;
    float m_run[8], l_run[8];
#pragma unroll
    for (int r = 0; r < 8; r++) { m_run[r] = -1e30f; l_run[r] = 0.0f; }

    const int nchunk = (t0_blk + 128) / CHUNK;  // uniform across the workgroup

    // ---- chunk stager ----
    auto stage = [&](int cidx, int bufIdx) {
#if USE_TDM
        if (wave == 0) {
            unsigned lds_off = (unsigned)(size_t)(&kc_lds[bufIdx][0]);
            unsigned long long ga =
                (unsigned long long)(size_t)(kbase + (size_t)cidx * CHUNK * 512);
            u32x4 g0;
            g0.x = 1u;                                  // count=1 valid descriptor
            g0.y = lds_off;                             // lds_addr
            g0.z = (unsigned)ga;                        // global_addr[31:0]
            g0.w = (unsigned)(ga >> 32) | (2u << 30);   // global_addr[56:32] | type=2
            i32x8 g1;
            g1[0] = (1 << 16)        // data_size = 2 bytes
                  | (1 << 20)        // pad_enable
                  | (7 << 22)        // pad_interval = 256 DWORDs (one 512-bf16 row)
                  | (3 << 25);       // pad_amount   = 4 DWORDs  (8 shorts)
            g1[1] = (512 << 16);     // tensor_dim0 = 512
            g1[2] = (CHUNK << 16);   // tensor_dim1 = 64
            g1[3] = (512 << 16);     // tile_dim0   = 512
            g1[4] = CHUNK;           // tile_dim1   = 64, tile_dim2 = 0
            g1[5] = 512;             // tensor_dim0_stride = 512
            g1[6] = 0;
            g1[7] = 0;
            __builtin_amdgcn_tensor_load_to_lds(g0, g1, (i32x4)0, (i32x4)0, (i32x8)0, 0);
        }
#else
        const uint4* src = (const uint4*)(kbase + (size_t)cidx * CHUNK * 512);
#pragma unroll
        for (int i = 0; i < 16; i++) {
            int e = tid + i * 256;       // 4096 uint4 = 64 rows x 64
            int row = e >> 6, c8 = e & 63;
            *(uint4*)&kc_lds[bufIdx][row * KPAD + c8 * 8] = src[e];
        }
#endif
    };

    stage(0, 0);  // prologue

    for (int ci = 0; ci < nchunk; ci++) {
        const int s0 = ci * CHUNK;
#if USE_TDM
        if (wave == 0) __builtin_amdgcn_s_wait_tensorcnt(0);
#endif
        __syncthreads();  // chunk ci visible; all waves done with buf[(ci+1)&1]
        if (ci + 1 < nchunk) stage(ci + 1, (ci + 1) & 1);  // overlap with compute

        if (s0 < t0 + 16) {  // wave-uniform: skip fully-masked chunks
            const unsigned short* kcur = &kc_lds[ci & 1][0];

            // ---- S = Q @ K^T (four 16-key subtiles, LDS-resident K) ----
            v8f s_acc[4];
#pragma unroll
            for (int st = 0; st < 4; st++)
#pragma unroll
                for (int e = 0; e < 8; e++) s_acc[st][e] = 0.0f;
#pragma unroll
            for (int st = 0; st < 4; st++) {
                const unsigned short* krow = kcur + (st * 16 + l16) * KPAD;
#pragma unroll
                for (int kk = 0; kk < 16; kk++) {
                    AFrag bf;
                    bf.u[0] = *(const uint4*)&krow[kk * 32 + half * 8];
                    bf.u[1] = *(const uint4*)&krow[kk * 32 + 16 + half * 8];
                    s_acc[st] = WMMA_BF16(qf[kk].v, bf.v, s_acc[st]);
                }
            }

            // ---- causal mask (diagonal-overlapping chunks) ----
            if (s0 + CHUNK > t0) {
#pragma unroll
                for (int r = 0; r < 8; r++) {
                    int t = t0 + r + half * 8;
#pragma unroll
                    for (int st = 0; st < 4; st++)
                        if (s0 + st * 16 + l16 > t) s_acc[st][r] = -1e30f;
                }
            }

            // ---- online softmax update ----
            float scale[8];
#pragma unroll
            for (int r = 0; r < 8; r++) {
                float mx = fmaxf(fmaxf(s_acc[0][r], s_acc[1][r]),
                                 fmaxf(s_acc[2][r], s_acc[3][r]));
#pragma unroll
                for (int mm = 1; mm < 16; mm <<= 1) mx = fmaxf(mx, __shfl_xor(mx, mm));
                float m_new = fmaxf(m_run[r], mx);
                float sc = __expf(m_run[r] - m_new);
                float p0 = __expf(s_acc[0][r] - m_new);
                float p1 = __expf(s_acc[1][r] - m_new);
                float p2 = __expf(s_acc[2][r] - m_new);
                float p3 = __expf(s_acc[3][r] - m_new);
                s_acc[0][r] = p0; s_acc[1][r] = p1; s_acc[2][r] = p2; s_acc[3][r] = p3;
                float sum = (p0 + p1) + (p2 + p3);
#pragma unroll
                for (int mm = 1; mm < 16; mm <<= 1) sum += __shfl_xor(sum, mm);
                l_run[r] = l_run[r] * sc + sum;
                m_run[r] = m_new;
                scale[r] = sc;
            }
#pragma unroll
            for (int j = 0; j < 32; j++)
#pragma unroll
                for (int r = 0; r < 8; r++) o[j][r] *= scale[r];

            // ---- P (C layout) -> two A fragments via per-wave LDS ----
#pragma unroll
            for (int r = 0; r < 8; r++) {
                int row = r + half * 8;
                unsigned pkA = pk_bf16(s_acc[0][r], s_acc[1][r]);
                unsigned pkB = pk_bf16(s_acc[2][r], s_acc[3][r]);
                p_lds[wave][row][l16]      = (unsigned short)pkA;
                p_lds[wave][row][16 + l16] = (unsigned short)(pkA >> 16);
                p_lds[wave][row][32 + l16] = (unsigned short)pkB;
                p_lds[wave][row][48 + l16] = (unsigned short)(pkB >> 16);
            }
            asm volatile("s_wait_dscnt 0x0" ::: "memory");
            AFrag pf0, pf1;
            pf0.u[0] = *(const uint4*)&p_lds[wave][l16][half * 8];
            pf0.u[1] = *(const uint4*)&p_lds[wave][l16][16 + half * 8];
            pf1.u[0] = *(const uint4*)&p_lds[wave][l16][32 + half * 8];
            pf1.u[1] = *(const uint4*)&p_lds[wave][l16][48 + half * 8];

            // ---- O += P @ V (V^T in global: contiguous s per v-column) ----
#pragma unroll
            for (int j = 0; j < 32; j++) {
                const unsigned short* vrow = vbase + (long)(j * 16 + l16) * 2048;
                AFrag vf0, vf1;
                vf0.u[0] = *(const uint4*)&vrow[s0 + half * 8];
                vf0.u[1] = *(const uint4*)&vrow[s0 + 16 + half * 8];
                vf1.u[0] = *(const uint4*)&vrow[s0 + 32 + half * 8];
                vf1.u[1] = *(const uint4*)&vrow[s0 + 48 + half * 8];
                o[j] = WMMA_BF16(pf0.v, vf0.v, o[j]);
                o[j] = WMMA_BF16(pf1.v, vf1.v, o[j]);
            }
        }
    }

    // ---- epilogue: O / l / sqrt(512) -> out[:, 1024:1536] ----
    const float inv_sqrtk = 0.044194173824159216f;  // 1/sqrt(512)
#pragma unroll
    for (int r = 0; r < 8; r++) {
        float inv = inv_sqrtk / l_run[r];
        float* orow = out + (long)(row_base + r + half * 8) * 1536 + 1024;
#pragma unroll
        for (int j = 0; j < 32; j++)
            orow[j * 16 + l16] = o[j][r] * inv;
    }
}

// ---------------------------------------------------------------------------
extern "C" void kernel_launch(void* const* d_in, const int* in_sizes, int n_in,
                              void* d_out, int out_size, void* d_ws, size_t ws_size,
                              hipStream_t stream) {
    const float* X  = (const float*)d_in[0];
    const float* Wk = (const float*)d_in[1];
    const float* bk = (const float*)d_in[2];
    const float* Wq = (const float*)d_in[3];
    const float* bq = (const float*)d_in[4];
    const float* Wv = (const float*)d_in[5];
    const float* bv = (const float*)d_in[6];
    float* out = (float*)d_out;

    unsigned short* qb  = (unsigned short*)d_ws;   // [8192,512] bf16
    unsigned short* kb  = qb + 8192 * 512;         // [8192,512] bf16
    unsigned short* vtb = kb + 8192 * 512;         // [4,512,2048] bf16

    qkv_proj_kernel<<<dim3(4, 128, 3), dim3(256), 0, stream>>>(
        X, Wq, bq, Wk, bk, Wv, bv, qb, kb, vtb);
    copy_passthrough<<<dim3(8192), dim3(256), 0, stream>>>(X, out);
    attn_kernel<<<dim3(16, 4), dim3(256), 0, stream>>>(qb, kb, vtb, out);
}
